// BertAdapterCapsuleMaskImp_51840255263217
// MI455X (gfx1250) — matrix operand order repaired and checked by reading, not compiled, not run on hardware
//
#include <hip/hip_runtime.h>
#include <hip/hip_bf16.h>
#include <math.h>

typedef __attribute__((ext_vector_type(16))) __bf16 v16bf;
typedef __attribute__((ext_vector_type(8)))  float  v8f;
typedef __attribute__((ext_vector_type(4)))  int    v4i;
typedef __attribute__((address_space(1))) v4i gv4i;    // global int4
typedef __attribute__((address_space(3))) v4i lv4i;    // LDS int4

#define Bb   128
#define Ss   128
#define Hh   768
#define Aa   2000
#define Tt   10
#define Cc   3
#define Mm   3
#define Ntok (Bb*Ss)     /* 16384 */
#define Apad 2048        /* A padded so K of GEMM2 is a multiple of 32 */
#define NEGV (-10000.0f)
#define EPSV (1e-16f)

#if defined(__HIP_DEVICE_COMPILE__) && __has_builtin(__builtin_amdgcn_global_load_async_to_lds_b128)
#define HAS_ASYNC_LDS 1
#else
#define HAS_ASYNC_LDS 0
#endif

__device__ __forceinline__ float gelu_exact(float x) {
    return 0.5f * x * (1.0f + erff(x * 0.70710678118654752440f));
}
__device__ __forceinline__ float sigmoidf_(float x) {
    return 1.0f / (1.0f + __expf(-x));
}

#if HAS_ASYNC_LDS
__device__ __forceinline__ void async_cp16(const __bf16* g, __bf16* l) {
    // GLOBAL_LOAD_ASYNC_TO_LDS_B128: per-lane 16B global -> LDS, tracked by ASYNCcnt
    __builtin_amdgcn_global_load_async_to_lds_b128((gv4i*)g, (lv4i*)l, 0, 0);
}
__device__ __forceinline__ void wait_async0() {
#if __has_builtin(__builtin_amdgcn_s_wait_asynccnt)
    __builtin_amdgcn_s_wait_asynccnt(0);
#else
    asm volatile("s_wait_asynccnt 0" ::: "memory");
#endif
}
#endif

// ---------------------------------------------------------------- gates ----
__global__ void k_gates(const float* __restrict__ s, const int* __restrict__ tptr,
                        const float* __restrict__ efc1, const float* __restrict__ efc2,
                        const float* __restrict__ elarger,
                        float* __restrict__ gfc1, float* __restrict__ gfc2,
                        float* __restrict__ glarger) {
    int i = blockIdx.x * blockDim.x + threadIdx.x;
    int t = tptr[0];
    float sv = s[0];
    if (i < Aa) gfc1[i] = sigmoidf_(sv * efc1[(size_t)t * Aa + i]);
    if (i < Hh) {
        gfc2[i]    = sigmoidf_(sv * efc2[(size_t)t * Hh + i]);
        glarger[i] = sigmoidf_(sv * elarger[(size_t)t * Hh + i]);
    }
}

// -------------------------------------------------- weight conversions ----
__global__ void k_conv_fc1(const float* __restrict__ w, __bf16* __restrict__ o) {
    int i = blockIdx.x * blockDim.x + threadIdx.x;          // over Apad*Hh
    if (i >= Apad * Hh) return;
    int a = i / Hh;
    o[i] = (a < Aa) ? (__bf16)w[i] : (__bf16)0.0f;
}
__global__ void k_conv_fc2(const float* __restrict__ w, __bf16* __restrict__ o) {
    int i = blockIdx.x * blockDim.x + threadIdx.x;          // over Hh*Apad
    if (i >= Hh * Apad) return;
    int h = i / Apad, a = i % Apad;
    o[i] = (a < Aa) ? (__bf16)w[(size_t)h * Aa + a] : (__bf16)0.0f;
}

// --------------------------------------- semantic capsules + routing ------
__global__ __launch_bounds__(128) void k_caps_route(
    const float* __restrict__ x, const int* __restrict__ tptr,
    const float* __restrict__ sem_w, const float* __restrict__ sem_b,
    const float* __restrict__ route_w, float* __restrict__ voteg) {
    __shared__ float xrow[Hh];
    __shared__ float sem30[Cc * Tt];
    __shared__ float pri[Mm * Tt * Cc];
    __shared__ float logits[Mm * Cc * Tt];   // [p = m*3+d][r]
    __shared__ float vote[Mm * Cc];
    int n = blockIdx.x, tid = threadIdx.x;
    int t = tptr[0];
    const float* xr = x + (size_t)n * Hh;
    for (int i = tid; i < Hh; i += 128) xrow[i] = xr[i];
    __syncthreads();
    if (tid < Cc * Tt) {
        int c = tid / Tt, r = tid % Tt;
        const float* wr = sem_w + ((size_t)r * Cc + c) * Hh;
        float acc = 0.f;
        for (int h = 0; h < Hh; ++h) acc = fmaf(xrow[h], wr[h], acc);
        sem30[c * Tt + r] = acc + sem_b[r * Cc + c];
    }
    __syncthreads();
    if (tid < Cc) {                          // squash over T per capsule c
        float sq = EPSV;
        for (int r = 0; r < Tt; ++r) { float v = sem30[tid * Tt + r]; sq += v * v; }
        float f = (sq / (1.f + sq)) * rsqrtf(sq);
        for (int r = 0; r < Tt; ++r) sem30[tid * Tt + r] *= f;
    }
    __syncthreads();
    if (tid < Mm * Tt * Cc) {                // priors via flat-reinterpreted x5
        int m = tid / (Tt * Cc), rem = tid % (Tt * Cc), r = rem / Cc, d = rem % Cc;
        float acc = 0.f;
        for (int c = 0; c < Cc; ++c)
            acc = fmaf(sem30[r * Cc + c],
                       route_w[(((size_t)m * Tt + r) * Cc + c) * Cc + d], acc);
        pri[(m * Tt + r) * Cc + d] = acc;
        logits[tid] = 0.f;
    }
    __syncthreads();
    for (int it = 0; it < 3; ++it) {
        if (tid < Mm * Cc) {
            int m = tid / Cc, d = tid % Cc;
            float l[Tt], mx = -1e30f;
            for (int r = 0; r < Tt; ++r) {
                float lv = logits[tid * Tt + r];
                lv = (r <= t) ? lv : NEGV;
                logits[tid * Tt + r] = lv;
                l[r] = lv; mx = fmaxf(mx, lv);
            }
            float se = 0.f;
            for (int r = 0; r < Tt; ++r) { l[r] = __expf(l[r] - mx); se += l[r]; }
            float inv = 1.f / se, v = 0.f;
            for (int r = 0; r < Tt; ++r) v = fmaf(l[r] * inv, pri[(m * Tt + r) * Cc + d], v);
            vote[tid] = v;
        }
        __syncthreads();
        if (it != 2 && tid < Mm * Tt) {
            int m = tid / Tt, r = tid % Tt;
            float sq = EPSV;
            for (int d = 0; d < Cc; ++d) { float v = vote[m * Cc + d]; sq += v * v; }
            float f = (sq / (1.f + sq)) * rsqrtf(sq);
            float delta = 0.f;
            for (int d = 0; d < Cc; ++d)
                delta = fmaf(pri[(m * Tt + r) * Cc + d], vote[m * Cc + d] * f, delta);
            for (int d = 0; d < Cc; ++d)
                logits[(m * Cc + d) * Tt + r] += delta;
        }
        __syncthreads();
    }
    if (tid < Mm * Cc)
        voteg[((size_t)(tid / Cc) * Ntok + n) * Cc + (tid % Cc)] = vote[tid];
}

// --------------- h = x + (h_caps @ larger_w^T + b) * glarger, as bf16 -----
__global__ void k_build_h(const float* __restrict__ x, const float* __restrict__ voteg,
                          const float* __restrict__ larger_w, const float* __restrict__ larger_b,
                          const float* __restrict__ glarger, __bf16* __restrict__ hbf) {
    int i = blockIdx.x * blockDim.x + threadIdx.x;
    if (i >= Ntok * Hh) return;
    int n = i / Hh, hc = i % Hh;
    const float* vv = voteg + (size_t)n * (Mm * Cc);
    const float* lw = larger_w + (size_t)hc * (Mm * Cc);
    float acc = larger_b[hc];
#pragma unroll
    for (int j = 0; j < Mm * Cc; ++j) acc = fmaf(vv[j], lw[j], acc);
    hbf[i] = (__bf16)(x[i] + acc * glarger[hc]);
}

// -------------------------------------------------- WMMA bf16 GEMM --------
// C[row, col] = sum_k A[row, k] * Bm[col, k]   (Bm is N-major, K-contiguous)
// Block tile 128(M) x 64(N), BK=32, 8 waves, each wave a 32x32 sub-tile.
// Double-buffered LDS; async global->LDS (ASYNCcnt) when available.
template <int MODE>
__global__ __launch_bounds__(256) void k_gemm_wmma(
    const __bf16* __restrict__ A, int lda,
    const __bf16* __restrict__ Bm, int ldb,
    int K, int nActual,
    const float* __restrict__ bias, const float* __restrict__ gate,
    const float* __restrict__ xres,
    __bf16* __restrict__ outB, float* __restrict__ outF) {
    __shared__ __bf16 lA[2][128 * 32];
    __shared__ __bf16 lB[2][64 * 32];
    int tid = threadIdx.x;
    int wid = tid >> 5, lane = tid & 31;
    int wm = wid & 3, wn = wid >> 2;           // 4x2 wave grid
    int half = lane >> 4, l16 = lane & 15;
    long rowBase = (long)blockIdx.y * 128;
    int  colBase = blockIdx.x * 64;
    int r = tid >> 2, cs = tid & 3;            // 64 rows x 4 16B segments

    const __bf16* gA1 = A + (rowBase + r) * (long)lda + cs * 8;
    const __bf16* gA2 = A + (rowBase + r + 64) * (long)lda + cs * 8;
    const __bf16* gB  = Bm + (long)(colBase + r) * ldb + cs * 8;
    int dO1 = r * 32 + cs * 8;
    int dO2 = (r + 64) * 32 + cs * 8;
    int KT = K >> 5;

    v8f acc[2][2] = {};
#if HAS_ASYNC_LDS
    async_cp16(gA1, &lA[0][dO1]);
    async_cp16(gA2, &lA[0][dO2]);
    async_cp16(gB,  &lB[0][dO1]);
#else
    uint4 rA1 = *(const uint4*)gA1;
    uint4 rA2 = *(const uint4*)gA2;
    uint4 rB  = *(const uint4*)gB;
#endif
    for (int kt = 0; kt < KT; ++kt) {
        int cur = kt & 1;
#if HAS_ASYNC_LDS
        wait_async0();                          // own copies for buf[cur] done
        __syncthreads();                        // all waves staged; buf[1-cur] free
        if (kt + 1 < KT) {                      // overlap DMA with WMMA below
            int nk = (kt + 1) << 5;
            async_cp16(gA1 + nk, &lA[1 - cur][dO1]);
            async_cp16(gA2 + nk, &lA[1 - cur][dO2]);
            async_cp16(gB + nk,  &lB[1 - cur][dO1]);
        }
#else
        *(uint4*)&lA[cur][dO1] = rA1;
        *(uint4*)&lA[cur][dO2] = rA2;
        *(uint4*)&lB[cur][dO1] = rB;
        __syncthreads();
        if (kt + 1 < KT) {                      // overlap global loads with WMMA
            int nk = (kt + 1) << 5;
            rA1 = *(const uint4*)(gA1 + nk);
            rA2 = *(const uint4*)(gA2 + nk);
            rB  = *(const uint4*)(gB + nk);
        }
#endif
        union { v16bf v; unsigned u[8]; } aF[2], bF[2];
#pragma unroll
        for (int sm = 0; sm < 2; ++sm) {        // A frag: 16x32, ISA layout
            const __bf16* base = &lA[cur][(wm * 32 + sm * 16 + l16) * 32];
#pragma unroll
            for (int v = 0; v < 8; ++v) {
                int kb = (v < 4) ? (2 * v + 8 * half) : (16 + 2 * (v - 4) + 8 * half);
                aF[sm].u[v] = *(const unsigned*)(base + kb);
            }
        }
#pragma unroll
        for (int sn = 0; sn < 2; ++sn) {        // B frag: 32x16, ISA layout
            const __bf16* base = &lB[cur][(wn * 32 + sn * 16 + l16) * 32];
#pragma unroll
            for (int v = 0; v < 8; ++v)
                bF[sn].u[v] = *(const unsigned*)(base + ((v < 4) ? (2 * v + 8 * half) : (16 + 2 * (v - 4) + 8 * half)));
        }
#pragma unroll
        for (int sm = 0; sm < 2; ++sm)
#pragma unroll
            for (int sn = 0; sn < 2; ++sn)
                acc[sm][sn] = __builtin_amdgcn_wmma_f32_16x16x32_bf16(
                    false, aF[sm].v, false, bF[sn].v, (short)0, acc[sm][sn], false, false);
    }
    // fused epilogue
#pragma unroll
    for (int sm = 0; sm < 2; ++sm)
#pragma unroll
        for (int sn = 0; sn < 2; ++sn)
#pragma unroll
            for (int v = 0; v < 8; ++v) {
                long row = rowBase + wm * 32 + sm * 16 + v + 8 * half;
                int  col = colBase + wn * 32 + sn * 16 + l16;
                float a = acc[sm][sn][v];
                if (MODE == 1) {
                    __bf16 o = (__bf16)0.0f;
                    if (col < nActual)
                        o = (__bf16)(gelu_exact(a + bias[col]) * gate[col]);
                    outB[row * Apad + col] = o;
                } else {
                    outF[row * Hh + col] =
                        xres[row * Hh + col] + gelu_exact(a + bias[col]) * gate[col];
                }
            }
}

// ------------------------------------------------------------- launch -----
extern "C" void kernel_launch(void* const* d_in, const int* in_sizes, int n_in,
                              void* d_out, int out_size, void* d_ws, size_t ws_size,
                              hipStream_t stream) {
    const float* x        = (const float*)d_in[0];
    const int*   tptr     = (const int*)d_in[1];
    const float* s        = (const float*)d_in[2];
    const float* fc1_w    = (const float*)d_in[3];
    const float* fc1_b    = (const float*)d_in[4];
    const float* fc2_w    = (const float*)d_in[5];
    const float* fc2_b    = (const float*)d_in[6];
    const float* efc1     = (const float*)d_in[7];
    const float* efc2     = (const float*)d_in[8];
    const float* sem_w    = (const float*)d_in[9];
    const float* sem_b    = (const float*)d_in[10];
    const float* route_w  = (const float*)d_in[11];
    const float* larger_w = (const float*)d_in[12];
    const float* larger_b = (const float*)d_in[13];
    const float* elarger  = (const float*)d_in[14];
    float* out = (float*)d_out;

    char* ws = (char*)d_ws;
    size_t off = 0;
    auto take = [&](size_t bytes) {
        char* p = ws + off;
        off = (off + bytes + 255) & ~(size_t)255;
        return p;
    };
    __bf16* fc1w_b  = (__bf16*)take((size_t)Apad * Hh * sizeof(__bf16));
    __bf16* fc2w_b  = (__bf16*)take((size_t)Hh * Apad * sizeof(__bf16));
    float*  gfc1    = (float*)take((size_t)Aa * sizeof(float));
    float*  gfc2    = (float*)take((size_t)Hh * sizeof(float));
    float*  glarger = (float*)take((size_t)Hh * sizeof(float));
    float*  voteg   = (float*)take((size_t)Mm * Ntok * Cc * sizeof(float));
    __bf16* hbf     = (__bf16*)take((size_t)Ntok * Hh * sizeof(__bf16));
    __bf16* h1bf    = (__bf16*)take((size_t)Ntok * Apad * sizeof(__bf16));
    (void)ws_size; (void)in_sizes; (void)n_in; (void)out_size;

    k_gates<<<(Apad + 255) / 256, 256, 0, stream>>>(s, tptr, efc1, efc2, elarger,
                                                    gfc1, gfc2, glarger);
    k_conv_fc1<<<(Apad * Hh + 255) / 256, 256, 0, stream>>>(fc1_w, fc1w_b);
    k_conv_fc2<<<(Hh * Apad + 255) / 256, 256, 0, stream>>>(fc2_w, fc2w_b);
    k_caps_route<<<Ntok, 128, 0, stream>>>(x, tptr, sem_w, sem_b, route_w, voteg);
    k_build_h<<<(Ntok * Hh + 255) / 256, 256, 0, stream>>>(x, voteg, larger_w, larger_b,
                                                           glarger, hbf);
    // GEMM1: [16384,768] x [2048(cols),768] -> gelu*gate -> h1 bf16 [16384,2048]
    k_gemm_wmma<1><<<dim3(Apad / 64, Ntok / 128), 256, 0, stream>>>(
        hbf, Hh, fc1w_b, Hh, Hh, Aa, fc1_b, gfc1, nullptr, h1bf, nullptr);
    // GEMM2: [16384,2048] x [768(cols),2048] -> x + gelu*gate -> out f32
    k_gemm_wmma<2><<<dim3(Hh / 64, Ntok / 128), 256, 0, stream>>>(
        h1bf, Apad, fc2w_b, Apad, Apad, Hh, fc2_b, gfc2, x, nullptr, out);
}
